// TinyRF_21268678050110
// MI455X (gfx1250) — compile-verified, matrix operand order checked by reading, not run on hardware
//
#include <hip/hip_runtime.h>
#include <hip/hip_bf16.h>
#include <stdint.h>

// Problem constants (match reference)
#define BATCH 16384
#define INF   64
#define OUTF  64
#define HD    1024
#define LAYERS 8
#define EPSV  1e-5f

typedef __attribute__((ext_vector_type(16))) __bf16    v16bf;
typedef __attribute__((ext_vector_type(8)))  float     v8f;
typedef __attribute__((ext_vector_type(4)))  uint32_t  v4u;

// Build a 16x32 bf16 WMMA operand fragment (8 VGPRs) from two 16B chunks.
// Per CDNA5 ISA 7.12.2 (16-bit A 16x32): lane m (0..15) holds K = k0+{0..7},
// k0+{16..23}; lanes 16..31 hold K = k0+{8..15}, k0+{24..31}. Caller applies
// the +8 element shift for the upper half-wave via 'ksel'.
union FragU { v4u q[2]; v16bf v; };
__device__ __forceinline__ v16bf load_frag(const uint16_t* p) {
  FragU f;
  f.q[0] = *reinterpret_cast<const v4u*>(p);        // k .. k+7
  f.q[1] = *reinterpret_cast<const v4u*>(p + 16);   // k+16 .. k+23
  return f.v;
}

__device__ __forceinline__ uint16_t f2bf(float f) {   // RNE f32 -> bf16
  uint32_t u = __builtin_bit_cast(uint32_t, f);
  u += 0x7fffu + ((u >> 16) & 1u);
  return (uint16_t)(u >> 16);
}
__device__ __forceinline__ float siluf(float x) { return x / (1.0f + __expf(-x)); }

__device__ __forceinline__ v8f wmma_bf16(v16bf a, v16bf b, v8f c) {
  return __builtin_amdgcn_wmma_f32_16x16x32_bf16(false, a, false, b, (short)0, c,
                                                 false, false);
}

// ---------------- prologue kernels ----------------

__global__ void k_f32_to_bf16(const float* __restrict__ src,
                              uint16_t* __restrict__ dst, int n) {
  int i = blockIdx.x * 256 + threadIdx.x;
  if (i < n) dst[i] = f2bf(src[i]);
}

// ta[b,j] = silu(t[b]*Wt[j] + bt[j])  (rank-1 "t embedding"), stored bf16
__global__ void k_ta(const float* __restrict__ t, const float* __restrict__ Wt,
                     const float* __restrict__ bt, uint16_t* __restrict__ ta) {
  int i = blockIdx.x * 256 + threadIdx.x;  // over BATCH*HD
  int b = i >> 10, j = i & (HD - 1);
  ta[i] = f2bf(siluf(t[b] * Wt[j] + bt[j]));
}

// h0[b,j] = sum_k x[b,k]*Wi[j,k] + bi[j]   (K=64, tiny; f32 VALU)
__global__ void k_h0(const float* __restrict__ x, const float* __restrict__ Wi,
                     const float* __restrict__ bi, float* __restrict__ h) {
  __shared__ float sx[INF];
  int b = blockIdx.x;
  if (threadIdx.x < INF) sx[threadIdx.x] = x[b * INF + threadIdx.x];
  __syncthreads();
  for (int i = 0; i < HD / 256; ++i) {
    int j = threadIdx.x + 256 * i;
    const float* w = Wi + j * INF;
    float acc = bi[j];
#pragma unroll
    for (int k = 0; k < INF; ++k) acc += sx[k] * w[k];
    h[(size_t)b * HD + j] = acc;
  }
}

// ---------------- per-layer kernel 1: LayerNorm + mod GEMM + build u ----------------
// One WG = 16 rows. 8 waves; wave w owns u-column-tiles [8w..8w+7]; for each it
// accumulates BOTH the shift tile (Wm rows n) and the scale tile (Wm rows H+n),
// so mod is never materialized: u = xn*(1+scale)+shift written as bf16.
// All B-fragment loads use ONE per-wave base pointer + compile-time-constant
// offsets (row j: j*16*HD elems = j*32KiB; scale block: +HD*HD elems = +2MiB;
// max ~2.33MiB < signed 24-bit instr offset) so the k-loop has no per-row
// 64-bit pointer updates.
__global__ __launch_bounds__(256) void k_mod_u(
    const float* __restrict__ h, const uint16_t* __restrict__ ta,
    const uint16_t* __restrict__ Wm, const float* __restrict__ bm,
    const float* __restrict__ gamma, const float* __restrict__ beta,
    uint16_t* __restrict__ u) {
  __shared__ float s_mu[16], s_rs[16];
  const int tid = threadIdx.x;
  const int wave = tid >> 5, lane = tid & 31;
  const int r0 = blockIdx.x * 16;

  // LayerNorm stats: each wave reduces 2 rows (wave32 shfl_xor tree)
  for (int rr = 0; rr < 2; ++rr) {
    int r = 2 * wave + rr;
    const float* hr = h + (size_t)(r0 + r) * HD;
    float s = 0.f, s2 = 0.f;
    for (int j = lane; j < HD; j += 32) { float v = hr[j]; s += v; s2 += v * v; }
#pragma unroll
    for (int m = 16; m >= 1; m >>= 1) {
      s  += __shfl_xor(s,  m, 32);
      s2 += __shfl_xor(s2, m, 32);
    }
    if (lane == 0) {
      float mu  = s  * (1.f / HD);
      float var = s2 * (1.f / HD) - mu * mu;
      s_mu[r] = mu;
      s_rs[r] = rsqrtf(var + EPSV);
    }
  }
  __syncthreads();

  const int mrow = lane & 15;
  const int ksel = (lane < 16) ? 0 : 8;
  const int ctb = wave * 8;

  const uint16_t* aBase = ta + (size_t)(r0 + mrow) * HD + ksel;
  // per-wave weight base: shift row of tile j=0 for this lane
  const uint16_t* wBase = Wm + (size_t)(ctb * 16 + mrow) * HD + ksel;

  v8f zero = {};
  v8f acc_s[8], acc_c[8];
#pragma unroll
  for (int j = 0; j < 8; ++j) { acc_s[j] = zero; acc_c[j] = zero; }

#pragma unroll 2
  for (int k0 = 0; k0 < HD; k0 += 32) {
    v16bf a = load_frag(aBase + k0);
#pragma unroll
    for (int j = 0; j < 8; ++j) {
      const uint16_t* ws = wBase + k0 + (size_t)j * (16 * HD);            // shift
      const uint16_t* wc = ws + (size_t)HD * HD;                          // scale
      __builtin_prefetch(ws + 32, 0, 3);  // next 64B line of this row, near cache
      __builtin_prefetch(wc + 32, 0, 3);
      v16bf bs = load_frag(ws);
      v16bf bc = load_frag(wc);
      acc_s[j] = wmma_bf16(a, bs, acc_s[j]);
      acc_c[j] = wmma_bf16(a, bc, acc_c[j]);
    }
  }

  // Epilogue: C/D layout — VGPR v, lanes 0-15: (M=v, N=lane); lanes 16-31: (M=v+8)
#pragma unroll
  for (int j = 0; j < 8; ++j) {
    int ncol = (ctb + j) * 16 + mrow;
    float g = gamma[ncol], be = beta[ncol];
    float b0 = bm[ncol], b1 = bm[HD + ncol];
#pragma unroll
    for (int v = 0; v < 8; ++v) {
      int m = (lane < 16) ? v : v + 8;
      size_t row = (size_t)(r0 + m);
      float hv = h[row * HD + ncol];
      float xn = (hv - s_mu[m]) * s_rs[m] * g + be;
      float shift = acc_s[j][v] + b0;
      float scale = acc_c[j][v] + b1;
      u[row * HD + ncol] = f2bf(xn * (1.f + scale) + shift);
    }
  }
}

// ---------------- per-layer kernel 2: y = silu(u @ Wf^T + bf); h += y ----------------
// One WG = 32 rows (2 M-tiles). Wave w owns N-tiles [8w..8w+7] for both M-tiles.
// Single A base (+32KiB const for 2nd M-tile) and single B base (+j*32KiB const).
__global__ __launch_bounds__(256) void k_ffn(
    const uint16_t* __restrict__ u, const uint16_t* __restrict__ Wf,
    const float* __restrict__ bfv, float* __restrict__ h) {
  const int tid = threadIdx.x;
  const int wave = tid >> 5, lane = tid & 31;
  const int r0 = blockIdx.x * 32;
  const int mrow = lane & 15;
  const int ksel = (lane < 16) ? 0 : 8;
  const int ntb = wave * 8;

  const uint16_t* aBase = u + (size_t)(r0 + mrow) * HD + ksel;
  const uint16_t* wBase = Wf + (size_t)(ntb * 16 + mrow) * HD + ksel;

  v8f zero = {};
  v8f acc[2][8];
#pragma unroll
  for (int j = 0; j < 8; ++j) { acc[0][j] = zero; acc[1][j] = zero; }

#pragma unroll 2
  for (int k0 = 0; k0 < HD; k0 += 32) {
    v16bf a0 = load_frag(aBase + k0);
    v16bf a1 = load_frag(aBase + k0 + (size_t)(16 * HD));
#pragma unroll
    for (int j = 0; j < 8; ++j) {
      const uint16_t* wr = wBase + k0 + (size_t)j * (16 * HD);
      __builtin_prefetch(wr + 32, 0, 3);
      v16bf b = load_frag(wr);
      acc[0][j] = wmma_bf16(a0, b, acc[0][j]);
      acc[1][j] = wmma_bf16(a1, b, acc[1][j]);
    }
  }

#pragma unroll
  for (int mt = 0; mt < 2; ++mt) {
#pragma unroll
    for (int j = 0; j < 8; ++j) {
      int col = (ntb + j) * 16 + mrow;
      float bb = bfv[col];
#pragma unroll
      for (int v = 0; v < 8; ++v) {
        int m = (lane < 16) ? v : v + 8;
        size_t row = (size_t)(r0 + mt * 16 + m);
        float y = siluf(acc[mt][j][v] + bb);
        h[row * HD + col] += y;   // f32 residual
      }
    }
  }
}

// ---------------- epilogue: out = h @ Wlast^T + blast (N=64, tiny) ----------------
__global__ void k_out(const float* __restrict__ h, const float* __restrict__ Wl,
                      const float* __restrict__ bl, float* __restrict__ out) {
  int i = blockIdx.x * 256 + threadIdx.x;  // over BATCH*OUTF
  int b = i >> 6, o = i & (OUTF - 1);
  const float4* hr = reinterpret_cast<const float4*>(h + (size_t)b * HD);
  const float4* wr = reinterpret_cast<const float4*>(Wl + (size_t)o * HD);
  float acc = bl[o];
  for (int k = 0; k < HD / 4; ++k) {
    float4 hv = hr[k], wv = wr[k];
    acc += hv.x * wv.x + hv.y * wv.y + hv.z * wv.z + hv.w * wv.w;
  }
  out[i] = acc;
}

// ---------------- host launcher ----------------
extern "C" void kernel_launch(void* const* d_in, const int* in_sizes, int n_in,
                              void* d_out, int out_size, void* d_ws, size_t ws_size,
                              hipStream_t stream) {
  const float* x     = (const float*)d_in[0];
  const float* t     = (const float*)d_in[1];
  const float* Wt    = (const float*)d_in[2];
  const float* bt    = (const float*)d_in[3];
  const float* Wi    = (const float*)d_in[4];
  const float* bi    = (const float*)d_in[5];
  const float* Wm    = (const float*)d_in[6];
  const float* bm    = (const float*)d_in[7];
  const float* Wf    = (const float*)d_in[8];
  const float* bfv   = (const float*)d_in[9];
  const float* gamma = (const float*)d_in[10];
  const float* beta  = (const float*)d_in[11];
  const float* Wlast = (const float*)d_in[12];
  const float* blast = (const float*)d_in[13];
  float* out = (float*)d_out;

  // Workspace layout (bytes):
  //   h   f32  : BATCH*HD*4          =  64 MiB
  //   u   bf16 : BATCH*HD*2          =  32 MiB
  //   ta  bf16 : BATCH*HD*2          =  32 MiB
  //   Wm  bf16 : L*2H*H*2            =  32 MiB
  //   Wf  bf16 : L*H*H*2             =  16 MiB
  char* ws = (char*)d_ws;
  const size_t BH = (size_t)BATCH * HD;
  float*    h   = (float*)ws;
  uint16_t* u   = (uint16_t*)(ws + BH * 4);
  uint16_t* tab = (uint16_t*)(ws + BH * 6);
  uint16_t* Wmb = (uint16_t*)(ws + BH * 8);
  uint16_t* Wfb = (uint16_t*)(ws + BH * 8 + (size_t)LAYERS * 2 * HD * HD * 2);

  const int nWm = LAYERS * 2 * HD * HD;  // 16,777,216
  const int nWf = LAYERS * HD * HD;      //  8,388,608
  k_f32_to_bf16<<<(nWm + 255) / 256, 256, 0, stream>>>(Wm, Wmb, nWm);
  k_f32_to_bf16<<<(nWf + 255) / 256, 256, 0, stream>>>(Wf, Wfb, nWf);
  k_ta<<<(int)(BH / 256), 256, 0, stream>>>(t, Wt, bt, tab);
  k_h0<<<BATCH, 256, 0, stream>>>(x, Wi, bi, h);

  for (int l = 0; l < LAYERS; ++l) {
    k_mod_u<<<BATCH / 16, 256, 0, stream>>>(
        h, tab, Wmb + (size_t)l * 2 * HD * HD, bm + (size_t)l * 2 * HD,
        gamma + (size_t)l * HD, beta + (size_t)l * HD, u);
    k_ffn<<<BATCH / 32, 256, 0, stream>>>(
        u, Wfb + (size_t)l * HD * HD, bfv + (size_t)l * HD, h);
  }

  k_out<<<BATCH * OUTF / 256, 256, 0, stream>>>(h, Wlast, blast, out);
}
